// VisionSelfAttention_82772609728633
// MI455X (gfx1250) — matrix operand order repaired and checked by reading, not compiled, not run on hardware
//
#include <hip/hip_runtime.h>

// ---------------------------------------------------------------------------
// VisionSelfAttention on MI455X (gfx1250), wave32 + WMMA.
//
// B=16, H=64, W=64, C=192, heads=8, d=32, ks=3.  NPIX = 65536.
// Pipeline:
//   1) prep: x -> bf16; pack Wq|Wk[:192]|Wv[:192] -> Wqkv_t [768][192] bf16;
//      Wo -> Wo_t [192][256] bf16; dist-bias tables dbk/dbv [9][256] f32
//      (fold dx*Wk[192] + dy*Wk[193] + bk, same for V).
//   2) QKV = Xb @ Wqkv  (v_wmma_f32_16x16x32_bf16), stored bf16 [65536][768].
//   3) local 3x3 attention, thread per (pixel, head), exact softmax. OOB
//      neighbors keep only the dist-bias K/V (matches zero-pad reference).
//   4) out = Ob @ Wo + bo (WMMA), f32.
// ---------------------------------------------------------------------------

typedef __attribute__((ext_vector_type(16))) __bf16 v16bf;
typedef __attribute__((ext_vector_type(8)))  float  v8f;

struct __attribute__((aligned(16))) U128 { unsigned int x, y, z, w; };
union FragAB { v16bf v; U128 q[2]; };

#define NPIX   65536
#define CDIM   192
#define HDDIM  256      // heads*key_dim
#define QKVN   768
#define SCALE  0.17677669529663687f   // 1/sqrt(32)

// ------------------------------ prep kernels -------------------------------

__global__ __launch_bounds__(256) void prep_convert_x(const float* __restrict__ x,
                                                      __bf16* __restrict__ xb, int n4) {
  int i = blockIdx.x * 256 + threadIdx.x;
  if (i < n4) {
    const float4 f = ((const float4*)x)[i];
    __bf16* o = xb + (size_t)i * 4;
    o[0] = (__bf16)f.x; o[1] = (__bf16)f.y; o[2] = (__bf16)f.z; o[3] = (__bf16)f.w;
  }
}

__global__ __launch_bounds__(256) void prep_pack_weights(const float* __restrict__ Wq,
                                                         const float* __restrict__ Wk,
                                                         const float* __restrict__ Wv,
                                                         const float* __restrict__ Wo,
                                                         __bf16* __restrict__ Wqkv_t,  // [768][192]
                                                         __bf16* __restrict__ Wo_t) {  // [192][256]
  int idx = blockIdx.x * 256 + threadIdx.x;
  const int T1 = QKVN * CDIM;            // 147456
  const int T2 = CDIM * HDDIM;           // 49152
  if (idx < T1) {
    int n = idx / CDIM, k = idx % CDIM;  // Wqkv_t[n][k]
    float v;
    if      (n < 256) v = Wq[k * 256 + n];           // Wq [192][256]
    else if (n < 512) v = Wk[k * 256 + (n - 256)];   // Wk [194][256], k<192
    else              v = Wv[k * 256 + (n - 512)];
    Wqkv_t[idx] = (__bf16)v;
  } else if (idx < T1 + T2) {
    int i2 = idx - T1;
    int c = i2 / 256, j = i2 % 256;      // Wo_t[c][j] = Wo[j][c], Wo [256][192]
    Wo_t[i2] = (__bf16)Wo[j * CDIM + c];
  }
}

__global__ __launch_bounds__(256) void prep_bias_tables(const float* __restrict__ Wk,
                                                        const float* __restrict__ bk,
                                                        const float* __restrict__ Wv,
                                                        const float* __restrict__ bv,
                                                        float* __restrict__ dbk,
                                                        float* __restrict__ dbv) {
  int idx = blockIdx.x * 256 + threadIdx.x;   // 9*256
  if (idx < 9 * HDDIM) {
    int p = idx >> 8, hd = idx & 255;
    float dx = (float)(p % 3 - 1);   // channel 192 of kv_in
    float dy = (float)(p / 3 - 1);   // channel 193
    dbk[idx] = dx * Wk[192 * 256 + hd] + dy * Wk[193 * 256 + hd] + bk[hd];
    dbv[idx] = dx * Wv[192 * 256 + hd] + dy * Wv[193 * 256 + hd] + bv[hd];
  }
}

// ------------------------------ WMMA GEMM ----------------------------------
// C[M][Ntot] = A[M][KTOT] (bf16, row-major) x Bt[Ntot][KTOT] (bf16, N-major).
// Block tile 128x64, 8 waves (4x2 grid), each wave 32x32 = 2x2 WMMA tiles.

template <int KTOT, bool OUT_BF16>
__global__ __launch_bounds__(256) void wmma_gemm(const __bf16* __restrict__ A,
                                                 const __bf16* __restrict__ Bt,
                                                 void* __restrict__ Cout,
                                                 const float* __restrict__ bias,
                                                 int Ntot) {
  constexpr int BM = 128, BN = 64, KC = 32;
  __shared__ __bf16 lsA[BM * KC];
  __shared__ __bf16 lsB[BN * KC];

  const int t    = threadIdx.x;
  const int wid  = t >> 5;
  const int lane = t & 31;
  const int wm   = wid & 3;        // 0..3 -> 32-row strip
  const int wn   = wid >> 2;       // 0..1 -> 32-col strip
  const int mBase = blockIdx.x * BM;
  const int nBase = blockIdx.y * BN;

  v8f acc[2][2] = {};

  const int arow = lane & 15;
  const int ahi  = lane >> 4;

  for (int k0 = 0; k0 < KTOT; k0 += KC) {
    // ---- stage A tile: 128 rows x 32 halfs, 2 threads per row (32B each)
    {
      const int row = t >> 1;
      const int c0  = (t & 1) * 16;
      const U128* src = (const U128*)(A + (size_t)(mBase + row) * KTOT + k0 + c0);
      U128* dst = (U128*)&lsA[row * KC + c0];
      dst[0] = src[0];
      dst[1] = src[1];
    }
    // ---- stage B tile (transposed layout [n][k]): 64 rows x 32 halfs
    {
      const int row = t >> 2;
      const int c0  = (t & 3) * 8;
      *(U128*)&lsB[row * KC + c0] =
          *(const U128*)(Bt + (size_t)(nBase + row) * KTOT + k0 + c0);
    }
    __syncthreads();

    // ---- fragments. A 16x32 bf16: lane<16 holds K0-7,16-23; lane>=16 K8-15,24-31.
    FragAB a[2], b[2];
#pragma unroll
    for (int i = 0; i < 2; ++i) {
      const int off = (wm * 32 + i * 16 + arow) * KC + ahi * 8;
      a[i].q[0] = *(const U128*)&lsA[off];
      a[i].q[1] = *(const U128*)&lsA[off + 16];
    }
    // B 32x16 bf16: lane n holds column n; lane<16 holds K0-7,16-23; lane>=16 K8-15,24-31.
#pragma unroll
    for (int j = 0; j < 2; ++j) {
      const int off = (wn * 32 + j * 16 + arow) * KC + ahi * 8;
      b[j].q[0] = *(const U128*)&lsB[off];
      b[j].q[1] = *(const U128*)&lsB[off + 16];
    }

#pragma unroll
    for (int i = 0; i < 2; ++i)
#pragma unroll
      for (int j = 0; j < 2; ++j)
        acc[i][j] = __builtin_amdgcn_wmma_f32_16x16x32_bf16(
            false, a[i].v, false, b[j].v, (short)0, acc[i][j], false, false);
    __syncthreads();
  }

  // ---- epilogue: VGPR r -> row = r + 8*(lane>=16), col = lane&15
  const int rlo = lane & 15;
  const int rhi = (lane >> 4) * 8;
#pragma unroll
  for (int i = 0; i < 2; ++i)
#pragma unroll
    for (int j = 0; j < 2; ++j) {
      const int col = nBase + wn * 32 + j * 16 + rlo;
#pragma unroll
      for (int r = 0; r < 8; ++r) {
        const int row = mBase + wm * 32 + i * 16 + rhi + r;
        const float v = acc[i][j][r];
        if (OUT_BF16)
          ((__bf16*)Cout)[(size_t)row * Ntot + col] = (__bf16)v;
        else
          ((float*)Cout)[(size_t)row * Ntot + col] = v + bias[col];
      }
    }
}

// --------------------------- local attention -------------------------------
// One thread per (pixel, head). QKV bf16 [NPIX][768]: q|k|v each 256 wide.

union Row32 { U128 q[4]; __bf16 h[32]; };

__device__ __forceinline__ void load_row32(Row32& r, const __bf16* p) {
  const U128* s = (const U128*)p;
  r.q[0] = s[0]; r.q[1] = s[1]; r.q[2] = s[2]; r.q[3] = s[3];
}

__global__ __launch_bounds__(256) void attn_local3x3(const __bf16* __restrict__ QKV,
                                                     const float* __restrict__ bq,
                                                     const float* __restrict__ dbk,
                                                     const float* __restrict__ dbv,
                                                     __bf16* __restrict__ O) {
  const int gid   = blockIdx.x * 256 + threadIdx.x;
  const int head  = gid & 7;
  const int pixel = gid >> 3;
  const int xw = pixel & 63;
  const int yh = (pixel >> 6) & 63;
  const int hd0 = head * 32;

  float q[32];
  {
    Row32 r;
    load_row32(r, QKV + (size_t)pixel * QKVN + hd0);
#pragma unroll
    for (int d = 0; d < 32; ++d) q[d] = (float)r.h[d] + bq[hd0 + d];
  }

  // scores over the 9 window slots
  float s[9];
  for (int p = 0; p < 9; ++p) {
    const int yy = yh + p / 3 - 1;
    const int xx = xw + p % 3 - 1;
    const bool inb = ((unsigned)yy < 64u) && ((unsigned)xx < 64u);
    const float* db = dbk + p * HDDIM + hd0;
    float acc = 0.f;
    if (inb) {
      const int np = pixel + (p / 3 - 1) * 64 + (p % 3 - 1);
      Row32 kr;
      load_row32(kr, QKV + (size_t)np * QKVN + HDDIM + hd0);
#pragma unroll
      for (int d = 0; d < 32; ++d) acc += q[d] * ((float)kr.h[d] + db[d]);
    } else {
#pragma unroll
      for (int d = 0; d < 32; ++d) acc += q[d] * db[d];
    }
    s[p] = acc * SCALE;
  }

  // softmax over 9
  float m = s[0];
#pragma unroll
  for (int p = 1; p < 9; ++p) m = fmaxf(m, s[p]);
  float sum = 0.f;
#pragma unroll
  for (int p = 0; p < 9; ++p) { s[p] = __expf(s[p] - m); sum += s[p]; }
  const float inv = 1.f / sum;

  // weighted V
  float o[32];
#pragma unroll
  for (int d = 0; d < 32; ++d) o[d] = 0.f;
  for (int p = 0; p < 9; ++p) {
    const float wgt = s[p] * inv;
    const int yy = yh + p / 3 - 1;
    const int xx = xw + p % 3 - 1;
    const bool inb = ((unsigned)yy < 64u) && ((unsigned)xx < 64u);
    const float* db = dbv + p * HDDIM + hd0;
    if (inb) {
      const int np = pixel + (p / 3 - 1) * 64 + (p % 3 - 1);
      Row32 vr;
      load_row32(vr, QKV + (size_t)np * QKVN + 2 * HDDIM + hd0);
#pragma unroll
      for (int d = 0; d < 32; ++d) o[d] += wgt * ((float)vr.h[d] + db[d]);
    } else {
#pragma unroll
      for (int d = 0; d < 32; ++d) o[d] += wgt * db[d];
    }
  }

  Row32 r;
#pragma unroll
  for (int d = 0; d < 32; ++d) r.h[d] = (__bf16)o[d];
  U128* dst = (U128*)(O + (size_t)pixel * HDDIM + hd0);
  dst[0] = r.q[0]; dst[1] = r.q[1]; dst[2] = r.q[2]; dst[3] = r.q[3];
}

// ------------------------------- launcher ----------------------------------

extern "C" void kernel_launch(void* const* d_in, const int* in_sizes, int n_in,
                              void* d_out, int out_size, void* d_ws, size_t ws_size,
                              hipStream_t stream) {
  const float* x  = (const float*)d_in[0];
  const float* Wq = (const float*)d_in[1];
  const float* bq = (const float*)d_in[2];
  const float* Wk = (const float*)d_in[3];
  const float* bk = (const float*)d_in[4];
  const float* Wv = (const float*)d_in[5];
  const float* bv = (const float*)d_in[6];
  const float* Wo = (const float*)d_in[7];
  const float* bo = (const float*)d_in[8];
  float* out = (float*)d_out;

  // workspace layout (all offsets 256B aligned)
  char* w = (char*)d_ws;
  size_t off = 0;
  __bf16* Xb     = (__bf16*)(w + off); off += (size_t)NPIX * CDIM * 2;   // 25.2 MB
  __bf16* Wqkv_t = (__bf16*)(w + off); off += (size_t)QKVN * CDIM * 2;   // 288 KB
  __bf16* Wo_t   = (__bf16*)(w + off); off += (size_t)CDIM * HDDIM * 2;  // 96 KB
  float*  dbk    = (float*) (w + off); off += 9 * HDDIM * 4;             // 9 KB
  float*  dbv    = (float*) (w + off); off += 9 * HDDIM * 4;             // 9 KB
  __bf16* QKV    = (__bf16*)(w + off); off += (size_t)NPIX * QKVN * 2;   // 100.7 MB
  __bf16* Ob     = (__bf16*)(w + off); off += (size_t)NPIX * HDDIM * 2;  // 33.6 MB
  (void)off; (void)ws_size; (void)in_sizes; (void)n_in; (void)out_size;

  // 1) prep
  {
    const int n4 = NPIX * CDIM / 4;
    prep_convert_x<<<(n4 + 255) / 256, 256, 0, stream>>>(x, Xb, n4);
    const int tw = QKVN * CDIM + CDIM * HDDIM;
    prep_pack_weights<<<(tw + 255) / 256, 256, 0, stream>>>(Wq, Wk, Wv, Wo, Wqkv_t, Wo_t);
    prep_bias_tables<<<9, 256, 0, stream>>>(Wk, bk, Wv, bv, dbk, dbv);
  }

  // 2) fused QKV projection: [65536,192] x [192,768] -> bf16
  wmma_gemm<CDIM, true><<<dim3(NPIX / 128, QKVN / 64), 256, 0, stream>>>(
      Xb, Wqkv_t, (void*)QKV, nullptr, QKVN);

  // 3) local 3x3 attention
  attn_local3x3<<<NPIX * 8 / 256, 256, 0, stream>>>(QKV, bq, dbk, dbv, Ob);

  // 4) output projection: [65536,256] x [256,192] + bo -> f32
  wmma_gemm<HDDIM, false><<<dim3(NPIX / 128, CDIM / 64), 256, 0, stream>>>(
      Ob, Wo_t, (void*)out, bo, CDIM);
}
